// TemporalDecayAttention_14448269984408
// MI455X (gfx1250) — matrix operand order repaired and checked by reading, not compiled, not run on hardware
//
#include <hip/hip_runtime.h>
#include <hip/hip_bf16.h>

#define D_MODEL 1024
#define N_QKV   3072
#define NHEAD   16
#define HEAD_DIM 64
#define SEQ_T   2048
#define BATCH   2
#define M_TOTAL (BATCH * SEQ_T)   // 4096

typedef unsigned short ushort_t;
typedef __attribute__((ext_vector_type(16))) __bf16 v16bf;
typedef __attribute__((ext_vector_type(8)))  float  v8f;

union FragAB { v16bf v; uint4 q[2]; };

__device__ __forceinline__ ushort_t f2bf(float f) {
  unsigned int u = __builtin_bit_cast(unsigned int, f);
  u += 0x7FFFu + ((u >> 16) & 1u);          // round-to-nearest-even
  return (ushort_t)(u >> 16);
}

// Async global->LDS 16-byte copy (CDNA5 GLOBAL_LOAD_ASYNC_TO_LDS_B128, ASYNCcnt).
// lds_dst low 32 bits of the flat address are the LDS byte address (ISA 10.2).
__device__ __forceinline__ void async_ld_b128(void* lds_dst, const void* gbase,
                                              unsigned byte_off) {
  unsigned lds_off = (unsigned)(size_t)lds_dst;
  asm volatile("global_load_async_to_lds_b128 %0, %1, %2"
               :: "v"(lds_off), "v"(byte_off),
                  "s"((unsigned long long)(size_t)gbase)
               : "memory");
}
__device__ __forceinline__ void wait_async0() {
  asm volatile("s_wait_asynccnt 0" ::: "memory");
}

// ---------------------------------------------------------------------------
// Kernel 1: qkv = x @ qkv_w^T + b   (M=4096, N=3072, K=1024), bf16 WMMA.
// q,k stored head-major [b][h][t][d]; v stored TRANSPOSED [b][h][d][t].
// ---------------------------------------------------------------------------
__global__ __launch_bounds__(256)
void qkv_gemm_kernel(const float* __restrict__ x, const float* __restrict__ qkv_w,
                     const float* __restrict__ qkv_b,
                     ushort_t* __restrict__ qbuf, ushort_t* __restrict__ kbuf,
                     ushort_t* __restrict__ vtbuf) {
  __shared__ ushort_t As[64 * 32];
  __shared__ ushort_t Bs[128 * 32];
  const int tid   = threadIdx.x;
  const int lane  = tid & 31;
  const int wid   = tid >> 5;
  const int m0    = (int)blockIdx.x * 64;
  const int n0    = (int)blockIdx.y * 128;
  const int msub  = wid & 3;
  const int nhalf = wid >> 2;
  const int hsel  = lane >> 4;
  const int nlane = lane & 15;
  v8f acc[4] = {};

  for (int k0 = 0; k0 < D_MODEL; k0 += 32) {
    __syncthreads();
#pragma unroll
    for (int e = 0; e < 8; ++e) {                       // A: 64x32 (convert f32->bf16)
      int idx = tid * 8 + e;
      int r = idx >> 5, c = idx & 31;
      As[idx] = f2bf(x[(size_t)(m0 + r) * D_MODEL + k0 + c]);
    }
#pragma unroll
    for (int e = 0; e < 16; ++e) {                      // B: W rows as-is (128x32)
      int idx = tid * 16 + e;
      int r = idx >> 5, c = idx & 31;
      Bs[idx] = f2bf(qkv_w[(size_t)(n0 + r) * D_MODEL + k0 + c]);
    }
    __syncthreads();

    FragAB a;
    {
      int kb = hsel * 8;
      const uint4* p = reinterpret_cast<const uint4*>(As);
      a.q[0] = p[((msub * 16 + nlane) * 32 + kb) >> 3];
      a.q[1] = p[((msub * 16 + nlane) * 32 + 16 + kb) >> 3];
    }
#pragma unroll
    for (int s = 0; s < 4; ++s) {
      FragAB b;
      int n_local = nhalf * 64 + s * 16 + nlane;
      int kb2 = hsel * 8;
      const uint4* p = reinterpret_cast<const uint4*>(Bs);
      b.q[0] = p[(n_local * 32 + kb2) >> 3];
      b.q[1] = p[(n_local * 32 + kb2 + 16) >> 3];
      acc[s] = __builtin_amdgcn_wmma_f32_16x16x32_bf16(
                   false, a.v, false, b.v, (short)0, acc[s], false, false);
    }
  }

#pragma unroll
  for (int s = 0; s < 4; ++s) {
    int n = n0 + nhalf * 64 + s * 16 + nlane;
    float bias = qkv_b[n];
    int which = n >> 10;        // 0=q 1=k 2=v
    int hd = n & 1023;
    int h = hd >> 6, d = hd & 63;
#pragma unroll
    for (int r = 0; r < 8; ++r) {
      int m = m0 + msub * 16 + r + hsel * 8;
      int b = m >> 11, t = m & 2047;
      ushort_t val = f2bf(acc[s][r] + bias);
      if (which == 2) {         // v transposed: [b][h][d][t]
        vtbuf[(((size_t)b * NHEAD + h) * HEAD_DIM + d) * SEQ_T + t] = val;
      } else {
        ushort_t* dst = (which == 0) ? qbuf : kbuf;
        dst[(((size_t)b * NHEAD + h) * SEQ_T + t) * HEAD_DIM + d] = val;
      }
    }
  }
}

// ---------------------------------------------------------------------------
// Kernel 2: flash attention with temporal decay bias.
// Grid (B*H, T/64), block 128 (4 waves); wave owns 16 q rows; 32-key chunks.
// K and V^T tiles staged via async global->LDS (no VGPR round-trip).
// ---------------------------------------------------------------------------
__global__ __launch_bounds__(128)
void attn_kernel(const ushort_t* __restrict__ qbuf, const ushort_t* __restrict__ kbuf,
                 const ushort_t* __restrict__ vtbuf, const float* __restrict__ decay,
                 ushort_t* __restrict__ attnbuf) {
  __shared__ ushort_t Ks[32 * 64];        // K chunk [tk][d]
  __shared__ ushort_t Vts[64 * 32];       // V chunk transposed [d][tk]
  __shared__ ushort_t Ps[4][16 * 32];     // per-wave P scratch
  const int tid   = threadIdx.x;
  const int lane  = tid & 31;
  const int wid   = tid >> 5;
  const int bh    = blockIdx.x;           // b*16+h
  const int h     = bh & 15;
  const int q0    = (int)blockIdx.y * 64 + wid * 16;
  const int hsel  = lane >> 4;
  const int nlane = lane & 15;
  const float alpha = log1pf(expf(decay[h]));   // softplus
  const float inv_scale = 0.125f;               // 1/sqrt(64)
  const ushort_t* kbase  = kbuf  + (size_t)bh * SEQ_T * HEAD_DIM;
  const ushort_t* vtbase = vtbuf + (size_t)bh * HEAD_DIM * SEQ_T;

  // Q fragments (two 32-wide K-dim halves of hd=64), loaded straight from global
  FragAB aq[2];
  {
    int kb = hsel * 8;
    const uint4* qp = reinterpret_cast<const uint4*>(
        qbuf + (size_t)bh * SEQ_T * HEAD_DIM + (size_t)(q0 + nlane) * HEAD_DIM);
    aq[0].q[0] = qp[(kb) >> 3];       aq[0].q[1] = qp[(16 + kb) >> 3];
    aq[1].q[0] = qp[(32 + kb) >> 3];  aq[1].q[1] = qp[(48 + kb) >> 3];
  }

  float mrow[8], lrow[8];
#pragma unroll
  for (int r = 0; r < 8; ++r) { mrow[r] = -1e30f; lrow[r] = 0.f; }
  v8f o[4] = {};

  for (int kt0 = 0; kt0 < SEQ_T; kt0 += 32) {
    __syncthreads();
    // K chunk (contiguous 32x64) + V^T chunk (64 rows of 32, stride T):
    // 2 async b128 per thread per tile.
#pragma unroll
    for (int ii = 0; ii < 2; ++ii) {
      int i = tid + ii * 128;                       // uint4 index 0..255
      async_ld_b128(&Ks[i * 8], kbase, (unsigned)((kt0 * 64 + i * 8) * 2));
      int d = i >> 2, c4 = i & 3;
      async_ld_b128(&Vts[i * 8], vtbase,
                    (unsigned)((d * SEQ_T + kt0 + c4 * 8) * 2));
    }
    wait_async0();
    __syncthreads();

    // S = Q @ K^T : two 16x16 key tiles, each accumulating over hd=64 (2 WMMAs)
    v8f sfrag[2];
#pragma unroll
    for (int s = 0; s < 2; ++s) {
      v8f c = {};
#pragma unroll
      for (int half = 0; half < 2; ++half) {
        FragAB b;
        int key = s * 16 + nlane;
        int kb2 = hsel * 8 + half * 32;
        const uint4* p = reinterpret_cast<const uint4*>(Ks);
        b.q[0] = p[(key * 64 + kb2) >> 3];
        b.q[1] = p[(key * 64 + kb2 + 16) >> 3];
        c = __builtin_amdgcn_wmma_f32_16x16x32_bf16(
                false, aq[half].v, false, b.v, (short)0, c, false, false);
      }
      sfrag[s] = c;
    }

    // online softmax with decay bias (row M lives in one 16-lane half)
    float p0[8], p1[8];
#pragma unroll
    for (int r = 0; r < 8; ++r) {
      int tq = q0 + r + hsel * 8;
      float l0 = sfrag[0][r] * inv_scale - alpha * fabsf((float)(tq - (kt0 + nlane)));
      float l1 = sfrag[1][r] * inv_scale - alpha * fabsf((float)(tq - (kt0 + 16 + nlane)));
      float mx = fmaxf(l0, l1);
#pragma unroll
      for (int off = 1; off < 16; off <<= 1) mx = fmaxf(mx, __shfl_xor(mx, off, 32));
      float mnew = fmaxf(mrow[r], mx);
      float e0 = __expf(l0 - mnew), e1 = __expf(l1 - mnew);
      float rs = e0 + e1;
#pragma unroll
      for (int off = 1; off < 16; off <<= 1) rs += __shfl_xor(rs, off, 32);
      float fac = __expf(mrow[r] - mnew);
      lrow[r] = lrow[r] * fac + rs;
      mrow[r] = mnew;
      p0[r] = e0; p1[r] = e1;
#pragma unroll
      for (int dsub = 0; dsub < 4; ++dsub) o[dsub][r] *= fac;
    }

    // P: C-layout -> bf16 A-layout via per-wave LDS (DS in-order within wave)
#pragma unroll
    for (int r = 0; r < 8; ++r) {
      int row = r + hsel * 8;
      Ps[wid][row * 32 + nlane]      = f2bf(p0[r]);
      Ps[wid][row * 32 + 16 + nlane] = f2bf(p1[r]);
    }
    asm volatile("s_wait_dscnt 0" ::: "memory");
    FragAB ap;
    {
      int kb = hsel * 8;
      const uint4* p = reinterpret_cast<const uint4*>(&Ps[wid][0]);
      ap.q[0] = p[(nlane * 32 + kb) >> 3];
      ap.q[1] = p[(nlane * 32 + 16 + kb) >> 3];
    }

    // O += P @ V : four d-subtiles, K-dim = 32 keys in one WMMA each
#pragma unroll
    for (int dsub = 0; dsub < 4; ++dsub) {
      FragAB b;
      int d = dsub * 16 + nlane;
      int kb2 = hsel * 8;
      const uint4* p = reinterpret_cast<const uint4*>(Vts);
      b.q[0] = p[(d * 32 + kb2) >> 3];
      b.q[1] = p[(d * 32 + kb2 + 16) >> 3];
      o[dsub] = __builtin_amdgcn_wmma_f32_16x16x32_bf16(
                    false, ap.v, false, b.v, (short)0, o[dsub], false, false);
    }
  }

  // normalize + store bf16 attn-out in [b*T+t][h*64+d] layout for the final GEMM
  const int b = bh >> 4;
#pragma unroll
  for (int r = 0; r < 8; ++r) {
    int t = q0 + r + hsel * 8;
    float inv_l = 1.0f / lrow[r];
#pragma unroll
    for (int dsub = 0; dsub < 4; ++dsub) {
      int d = dsub * 16 + nlane;
      attnbuf[(size_t)(b * SEQ_T + t) * D_MODEL + h * HEAD_DIM + d] = f2bf(o[dsub][r] * inv_l);
    }
  }
}

// ---------------------------------------------------------------------------
// Kernel 3: out = attn @ out_w^T + out_b  (M=4096, N=1024, K=1024), f32 output.
// bf16 A tile staged async (already bf16); W tile converted through VGPRs.
// ---------------------------------------------------------------------------
__global__ __launch_bounds__(256)
void out_gemm_kernel(const ushort_t* __restrict__ attnbuf, const float* __restrict__ out_w,
                     const float* __restrict__ out_b, float* __restrict__ out) {
  __shared__ ushort_t As[64 * 32];
  __shared__ ushort_t Bs[128 * 32];
  const int tid = threadIdx.x;
  const int lane = tid & 31;
  const int wid = tid >> 5;
  const int m0 = (int)blockIdx.x * 64;
  const int n0 = (int)blockIdx.y * 128;
  const int msub = wid & 3;
  const int nhalf = wid >> 2;
  const int hsel = lane >> 4;
  const int nlane = lane & 15;
  v8f acc[4] = {};

  for (int k0 = 0; k0 < D_MODEL; k0 += 32) {
    __syncthreads();
    // A tile: one async b128 per thread (8 contiguous bf16)
    async_ld_b128(&As[tid * 8], attnbuf,
                  (unsigned)((((m0 + (tid >> 2)) * D_MODEL) + k0 + (tid & 3) * 8) * 2));
#pragma unroll
    for (int e = 0; e < 16; ++e) {
      int idx = tid * 16 + e;
      int r = idx >> 5, c = idx & 31;
      Bs[idx] = f2bf(out_w[(size_t)(n0 + r) * D_MODEL + k0 + c]);
    }
    wait_async0();
    __syncthreads();

    FragAB a;
    {
      int kb = hsel * 8;
      const uint4* p = reinterpret_cast<const uint4*>(As);
      a.q[0] = p[((msub * 16 + nlane) * 32 + kb) >> 3];
      a.q[1] = p[((msub * 16 + nlane) * 32 + 16 + kb) >> 3];
    }
#pragma unroll
    for (int s = 0; s < 4; ++s) {
      FragAB b;
      int n_local = nhalf * 64 + s * 16 + nlane;
      int kb2 = hsel * 8;
      const uint4* p = reinterpret_cast<const uint4*>(Bs);
      b.q[0] = p[(n_local * 32 + kb2) >> 3];
      b.q[1] = p[(n_local * 32 + kb2 + 16) >> 3];
      acc[s] = __builtin_amdgcn_wmma_f32_16x16x32_bf16(
                   false, a.v, false, b.v, (short)0, acc[s], false, false);
    }
  }

#pragma unroll
  for (int s = 0; s < 4; ++s) {
    int n = n0 + nhalf * 64 + s * 16 + nlane;
    float bias = out_b[n];
#pragma unroll
    for (int r = 0; r < 8; ++r) {
      int m = m0 + msub * 16 + r + hsel * 8;
      out[(size_t)m * D_MODEL + n] = acc[s][r] + bias;
    }
  }
}

// ---------------------------------------------------------------------------
extern "C" void kernel_launch(void* const* d_in, const int* in_sizes, int n_in,
                              void* d_out, int out_size, void* d_ws, size_t ws_size,
                              hipStream_t stream) {
  (void)in_sizes; (void)n_in; (void)out_size; (void)ws_size;
  const float* x      = (const float*)d_in[0];
  const float* qkv_w  = (const float*)d_in[1];
  const float* qkv_b  = (const float*)d_in[2];
  const float* out_w  = (const float*)d_in[3];
  const float* out_b  = (const float*)d_in[4];
  const float* decay  = (const float*)d_in[5];
  float* out = (float*)d_out;

  const size_t buf_elems = (size_t)M_TOTAL * D_MODEL;   // 4M bf16 each
  ushort_t* qbuf  = (ushort_t*)d_ws;
  ushort_t* kbuf  = qbuf + buf_elems;
  ushort_t* vtbuf = kbuf + buf_elems;
  ushort_t* abuf  = vtbuf + buf_elems;

  qkv_gemm_kernel<<<dim3(M_TOTAL / 64, N_QKV / 128), 256, 0, stream>>>(
      x, qkv_w, qkv_b, qbuf, kbuf, vtbuf);
  attn_kernel<<<dim3(BATCH * NHEAD, SEQ_T / 64), 128, 0, stream>>>(
      qbuf, kbuf, vtbuf, decay, abuf);
  out_gemm_kernel<<<dim3(M_TOTAL / 64, D_MODEL / 128), 256, 0, stream>>>(
      abuf, out_w, out_b, out);
}
